// GemLiteLinearCUDA_49718541418970
// MI455X (gfx1250) — matrix operand order repaired and checked by reading, not compiled, not run on hardware
//
#include <hip/hip_runtime.h>
#include <hip/hip_fp16.h>

#define N_OUT 8192
#define K_IN  8192
#define GROUP 128
#define BATCH 4

typedef _Float16 v16h __attribute__((ext_vector_type(16)));
typedef float    v8f  __attribute__((ext_vector_type(8)));

union V16H { v16h v; uint4 q[2]; unsigned u[8]; };
union H2U  { __half2 h; unsigned u; };

// One wave computes a 16(n) x 16(batch-padded) f32 tile over a K-slice.
// A = dequantized 4-bit W (16x32 f16), B = x^T (32x16 f16, cols>=4 read a zero pad).
__global__ __launch_bounds__(128)
void gemlite_wmma_kernel(const __half* __restrict__ x,
                         const int*    __restrict__ Wp,
                         const __half* __restrict__ scales,
                         const __half* __restrict__ zeros,
                         float*        __restrict__ ws,   // [SPLIT][BATCH][N] partials, or null
                         __half*       __restrict__ out,  // direct f16 out when ws==null
                         int KS)                          // K-slice per block
{
  extern __shared__ __half smem[];          // BATCH*KS halves of x slice + 256-half zero pad
  const int tid  = threadIdx.x;
  const int lane = tid & 31;
  const int wave = tid >> 5;
  const int h    = lane >> 4;               // lane half (K sub-pattern)
  const int bcol = lane & 15;               // batch column of B/D (valid < 4)
  const int bx   = blockIdx.x;
  const int by   = blockIdx.y;              // K-split index
  const int n0   = bx * 64 + wave * 16;
  const int n    = n0 + (lane & 15);        // A-matrix row for this lane
  const int padoff = BATCH * KS;            // zero pad (256 halves)

  // ---- stage x K-slice into LDS (b128), plus zero pad for invalid B columns ----
  {
    const int chunksPerRow = KS >> 3;       // uint4 = 8 halves
    for (int idx = tid; idx < BATCH * chunksPerRow; idx += 128) {
      const int b  = idx / chunksPerRow;
      const int kc = idx - b * chunksPerRow;
      uint4 v = *(const uint4*)(x + b * K_IN + by * KS + kc * 8);
      *(uint4*)(smem + b * KS + kc * 8) = v;
    }
    if (tid < 32) { uint4 z = {0u,0u,0u,0u}; *(uint4*)(smem + padoff + tid * 8) = z; }
  }
  __syncthreads();

  const bool valid    = (bcol < BATCH);
  const int  xrowbase = bcol * KS + h * 8;           // valid-lane base (halves)
  const int* wrow = Wp + n * (K_IN / 8);
  const __half* srow = scales + n * (K_IN / GROUP);
  const __half* zrow = zeros  + n * (K_IN / GROUP);

  v8f acc = {};
  const int nblk = KS >> 8;                 // 256-K blocks in this slice
  const int blk0 = (by * KS) >> 8;          // first global 256-K block

  // ---- preload block 0 (weights + group meta), double-buffered across loop ----
  const int* wp0 = wrow + blk0 * 32 + h * 8;
  uint4 qa = *(const uint4*)(wp0);
  uint4 qb = *(const uint4*)(wp0 + 4);
  uint4 qc = *(const uint4*)(wp0 + 16);
  uint4 qd = *(const uint4*)(wp0 + 20);
  unsigned smeta = *(const unsigned*)(srow + blk0 * 2);
  unsigned zmeta = *(const unsigned*)(zrow + blk0 * 2);

  for (int b8 = 0; b8 < nblk; ++b8) {
    // consume current block's registers
    const unsigned w[16] = { qa.x,qa.y,qa.z,qa.w, qb.x,qb.y,qb.z,qb.w,
                             qc.x,qc.y,qc.z,qc.w, qd.x,qd.y,qd.z,qd.w };
    H2U su; su.u = smeta;
    H2U zu; zu.u = zmeta;

    // issue next block's global loads early (hidden under 8 chunks of dequant+WMMA)
    if (b8 + 1 < nblk) {
      const int* wpn = wrow + (blk0 + b8 + 1) * 32 + h * 8;
      qa = *(const uint4*)(wpn);
      qb = *(const uint4*)(wpn + 4);
      qc = *(const uint4*)(wpn + 16);
      qd = *(const uint4*)(wpn + 20);
      smeta = *(const unsigned*)(srow + (blk0 + b8 + 1) * 2);
      zmeta = *(const unsigned*)(zrow + (blk0 + b8 + 1) * 2);
    }

    // dequant constants for the two groups of this 256-K block
    const float s0f = __half2float(__low2half(su.h));
    const float s1f = __half2float(__high2half(su.h));
    const float z0f = __half2float(__low2half(zu.h));
    const float z1f = __half2float(__high2half(zu.h));
    // magic dequant: (0x6400|nib) == 1024+nib exactly; val = fma(h, s, -(1024+z)*s)
    const __half2 sc[2] = { __float2half2_rn(s0f), __float2half2_rn(s1f) };
    const __half2 bi[2] = { __float2half2_rn(-(1024.f + z0f) * s0f),
                            __float2half2_rn(-(1024.f + z1f) * s1f) };

    // loop-invariant LDS base for this block: all chunk loads use immediate offsets
    const __half* xb = smem + (valid ? (xrowbase + b8 * 256) : padoff);

    // pipeline B fragments one chunk ahead
    V16H bcur;
    bcur.q[0] = *(const uint4*)(xb);
    bcur.q[1] = *(const uint4*)(xb + 16);

#pragma unroll
    for (int c = 0; c < 8; ++c) {           // 8 WMMAs of K=32 per 256-K block
      V16H bnxt;
      if (c < 7) {
        bnxt.q[0] = *(const uint4*)(xb + (c + 1) * 32);
        bnxt.q[1] = *(const uint4*)(xb + (c + 1) * 32 + 16);
      }
      const int sh  = 28 - 4 * c;           // nibble shift for sub-chunk c
      const int grp = c >> 2;
      V16H a;
#pragma unroll
      for (int p = 0; p < 8; ++p) {         // packed dequant: 2 elems / v_pk_fma_f16
        unsigned lo = (w[2*p]   >> sh) & 0xFu;
        unsigned hi = (w[2*p+1] >> sh) & 0xFu;
        H2U t; t.u = 0x64006400u | lo | (hi << 16);
        H2U r; r.h = __hfma2(t.h, sc[grp], bi[grp]);
        a.u[p] = r.u;
      }
      acc = __builtin_amdgcn_wmma_f32_16x16x32_f16(false, a.v, false, bcur.v,
                                                   (short)0, acc, false, false);
      if (c < 7) bcur = bnxt;
    }
  }

  // D layout: lane holds col=bcol, rows n0+8h .. n0+8h+7 (consecutive n)
  const int nout = n0 + 8 * h;
  if (valid) {
    if (ws) {
      float* p = ws + ((size_t)(by * BATCH + bcol)) * N_OUT + nout;
      float4 f0 = { acc[0], acc[1], acc[2], acc[3] };
      float4 f1 = { acc[4], acc[5], acc[6], acc[7] };
      *(float4*)(p)     = f0;
      *(float4*)(p + 4) = f1;
    } else {
      __half hv[8];
#pragma unroll
      for (int r = 0; r < 8; ++r) hv[r] = __float2half(acc[r]);
      *(uint4*)(out + (size_t)bcol * N_OUT + nout) = *(const uint4*)hv;
    }
  }
}

__global__ __launch_bounds__(256)
void gemlite_reduce_kernel(const float* __restrict__ ws, __half* __restrict__ out,
                           int split, int total)
{
  int i = blockIdx.x * 256 + threadIdx.x;
  if (i < total) {
    float s = 0.f;
    for (int p = 0; p < split; ++p) s += ws[(size_t)p * total + i];
    out[i] = __float2half(s);
  }
}

extern "C" void kernel_launch(void* const* d_in, const int* in_sizes, int n_in,
                              void* d_out, int out_size, void* d_ws, size_t ws_size,
                              hipStream_t stream) {
  (void)in_sizes; (void)n_in; (void)out_size;
  const __half* x      = (const __half*)d_in[0];
  const int*    Wp     = (const int*)   d_in[1];
  const __half* scales = (const __half*)d_in[2];
  const __half* zeros  = (const __half*)d_in[3];
  __half* out = (__half*)d_out;

  const int SPLIT = 4;
  const size_t need = (size_t)SPLIT * BATCH * N_OUT * sizeof(float);

  if (ws_size >= need && d_ws) {
    // K-split x4 for occupancy: 512 blocks x 4 waves; f32 partials in d_ws.
    const int KS = K_IN / SPLIT;                       // 2048
    const size_t lds = (size_t)(BATCH * KS + 256) * sizeof(__half);  // ~16.5 KB
    dim3 grid(N_OUT / 64, SPLIT);
    gemlite_wmma_kernel<<<grid, 128, lds, stream>>>(x, Wp, scales, zeros,
                                                    (float*)d_ws, nullptr, KS);
    const int total = BATCH * N_OUT;
    gemlite_reduce_kernel<<<(total + 255) / 256, 256, 0, stream>>>(
        (const float*)d_ws, out, SPLIT, total);
  } else {
    // Fallback: single pass over full K, direct fp16 store (no workspace).
    const int KS = K_IN;
    const size_t lds = (size_t)(BATCH * KS + 256) * sizeof(__half);  // ~64.5 KB
    dim3 grid(N_OUT / 64, 1);
    gemlite_wmma_kernel<<<grid, 128, lds, stream>>>(x, Wp, scales, zeros,
                                                    nullptr, out, KS);
  }
}
